// Decoder_45509473468804
// MI455X (gfx1250) — matrix operand order repaired and checked by reading, not compiled, not run on hardware
//
#include <hip/hip_runtime.h>
#include <hip/hip_bf16.h>
#include <stdint.h>

// ---------------------------------------------------------------------------
// Edge-MLP decoder, fused, bf16 WMMA (gfx1250 / CDNA5, wave32).
//
//   h  = concat(x[src], x[dst])           [E,64]
//   h1 = relu(h  @ W1 + b1)               [E,32]
//   h2 = relu(h1 @ W2 + b2)               [E,32]
//   h3 = relu(h2 @ W3 + b3)               [E,32]
//   h4 = h3 @ W4 + b4                     [E,1]
//   out = LN(h4) over dim of size 1  ==  computed literally
//
// One wave == one 16-edge tile. 8 x v_wmma_f32_16x16x32_bf16 per tile.
// ---------------------------------------------------------------------------

#define HID 32
#define LN_EPS 1e-5f

typedef __attribute__((ext_vector_type(16))) __bf16          v16bf;
typedef __attribute__((ext_vector_type(16))) unsigned short  v16u;
typedef __attribute__((ext_vector_type(8)))  float           v8f;

static __device__ __forceinline__ unsigned short f2bfbits(float f) {
    __bf16 h = (__bf16)f;
    return __builtin_bit_cast(unsigned short, h);
}

// ---------------------------------------------------------------------------
// Pre-pack W1 (64x32), W2 (32x32), W3 (32x32) into bf16 B-matrix lane layout.
// B tile = 32x16 (KxN). Per-lane vector of 16 bf16:
//   lane = khalf*16 + n   (khalf selects K 0..7,16..23 vs 8..15,24..31 of the chunk)
//   elem j = W[kbase + khalf*8 + j + (j >= 8 ? 8 : 0)][ntile*16 + n]
// ws layout (bf16 elems): 8 tiles x 512:
//   tiles 0..3 : W1 [kchunk][ntile],  tiles 4..5 : W2 [ntile],  6..7 : W3
// ---------------------------------------------------------------------------
__global__ void prep_weights(const float* __restrict__ W1,
                             const float* __restrict__ W2,
                             const float* __restrict__ W3,
                             unsigned short* __restrict__ ws) {
    int t = blockIdx.x * blockDim.x + threadIdx.x;
    if (t >= 8 * 512) return;
    int tile = t >> 9;        // 0..7
    int idx  = t & 511;       // lane*16 + j
    int lane = idx >> 4;
    int j    = idx & 15;
    int khalf = lane >> 4;
    int n     = lane & 15;
    const float* W;
    int kbase, ntile;
    if (tile < 4)      { W = W1; kbase = (tile >> 1) * 32; ntile = tile & 1; }
    else if (tile < 6) { W = W2; kbase = 0;                ntile = tile - 4; }
    else               { W = W3; kbase = 0;                ntile = tile - 6; }
    int k = kbase + khalf * 8 + j + ((j >= 8) ? 8 : 0);
    ws[t] = f2bfbits(W[k * HID + ntile * 16 + n]);
}

static __device__ __forceinline__ v16bf load_btile(const unsigned short* wt,
                                                   int tile, int lane) {
    v16u raw = ((const v16u*)wt)[tile * 32 + lane];
    return __builtin_bit_cast(v16bf, raw);
}

static __device__ __forceinline__ v8f wmma_bf16(v16bf a, v16bf b, v8f c) {
    return __builtin_amdgcn_wmma_f32_16x16x32_bf16(
        /*neg_a=*/false, a, /*neg_b=*/false, b,
        /*c_mod=*/(short)0, c, /*reuse_a=*/false, /*reuse_b=*/false);
}

// bias + relu on C-layout accumulators, scatter to LDS as H[16][32] (row-major)
static __device__ __forceinline__ void relu_bias_to_lds(v8f c0, v8f c1,
                                                        float bias0, float bias1,
                                                        float* S, int r, int half) {
#pragma unroll
    for (int v = 0; v < 8; ++v) {
        int m = v + 8 * half;                 // C layout: M = vgpr + 8*(lane>=16)
        S[m * 32 + r]      = fmaxf(c0[v] + bias0, 0.f);   // N = r        (tile 0)
        S[m * 32 + 16 + r] = fmaxf(c1[v] + bias1, 0.f);   // N = 16 + r   (tile 1)
    }
    __builtin_amdgcn_wave_barrier();          // LDS is in-order per wave on CDNA5
}

// gather 16x32 f32 row-major from LDS into bf16 A-matrix layout
static __device__ __forceinline__ v16bf lds_to_amat(const float* S, int r, int half) {
    v16bf a;
    int kb = half * 8;                        // lanes16-31 hold K 8..15 / 24..31
#pragma unroll
    for (int j = 0; j < 8; ++j) {
        a[j]     = (__bf16)S[r * 32 + kb + j];
        a[8 + j] = (__bf16)S[r * 32 + 16 + kb + j];
    }
    __builtin_amdgcn_wave_barrier();
    return a;
}

__global__ __launch_bounds__(256) void edge_mlp(
    const float*          __restrict__ x_node,
    const int64_t*        __restrict__ edge_index,   // [2, E] int64
    const float*          __restrict__ b1,
    const float*          __restrict__ b2,
    const float*          __restrict__ b3,
    const float*          __restrict__ W4,           // [32,1]
    const float*          __restrict__ b4,           // [1]
    const float*          __restrict__ ln_gamma,     // [1]
    const float*          __restrict__ ln_beta,      // [1]
    const unsigned short* __restrict__ wtiles,       // packed bf16 B tiles
    float*                __restrict__ out,
    int nEdges) {
    __shared__ float smem[8][16 * 32];                // 2 KB per wave
    const int lane = threadIdx.x & 31;
    const int wid  = threadIdx.x >> 5;
    const int tile = blockIdx.x * 8 + wid;
    const long tbase = (long)tile * 16;
    const int r    = lane & 15;                       // tile row (edge within tile)
    const int half = lane >> 4;
    float* S = smem[wid];

    long e = tbase + r;
    if (e >= nEdges) e = nEdges - 1;                  // clamp: keep EXEC all-ones
    const int64_t srcI = edge_index[e];
    const int64_t dstI = edge_index[(long)nEdges + e];
    const float* ps = x_node + (long)srcI * HID;
    const float* pd = x_node + (long)dstI * HID;

    // ---- A matrices for layer 1: chunk0 = src feats (K 0..31), chunk1 = dst
    v16bf a0, a1;
    {
        int kb = half * 8;
#pragma unroll
        for (int j = 0; j < 8; ++j) {
            a0[j]     = (__bf16)ps[kb + j];
            a0[8 + j] = (__bf16)ps[16 + kb + j];
            a1[j]     = (__bf16)pd[kb + j];
            a1[8 + j] = (__bf16)pd[16 + kb + j];
        }
    }

    // ---- layer 1: [16x64] @ [64x32] -> 4 WMMAs
    v8f c0 = {}, c1 = {};
    c0 = wmma_bf16(a0, load_btile(wtiles, 0, lane), c0);   // k0, n0
    c0 = wmma_bf16(a1, load_btile(wtiles, 2, lane), c0);   // k1, n0
    c1 = wmma_bf16(a0, load_btile(wtiles, 1, lane), c1);   // k0, n1
    c1 = wmma_bf16(a1, load_btile(wtiles, 3, lane), c1);   // k1, n1
    relu_bias_to_lds(c0, c1, b1[r], b1[16 + r], S, r, half);

    // ---- layer 2: [16x32] @ [32x32] -> 2 WMMAs
    v16bf a2 = lds_to_amat(S, r, half);
    c0 = (v8f){}; c1 = (v8f){};
    c0 = wmma_bf16(a2, load_btile(wtiles, 4, lane), c0);
    c1 = wmma_bf16(a2, load_btile(wtiles, 5, lane), c1);
    relu_bias_to_lds(c0, c1, b2[r], b2[16 + r], S, r, half);

    // ---- layer 3: [16x32] @ [32x32] -> 2 WMMAs
    v16bf a3 = lds_to_amat(S, r, half);
    c0 = (v8f){}; c1 = (v8f){};
    c0 = wmma_bf16(a3, load_btile(wtiles, 6, lane), c0);
    c1 = wmma_bf16(a3, load_btile(wtiles, 7, lane), c1);
    relu_bias_to_lds(c0, c1, b3[r], b3[16 + r], S, r, half);

    // ---- layer 4 (32 -> 1) + LayerNorm over dim of size 1, lanes 0..15
    if (half == 0) {
        float acc = 0.f;
#pragma unroll
        for (int j = 0; j < HID; ++j) acc = fmaf(S[r * 32 + j], W4[j], acc);
        float h    = acc + b4[0];
        float mu   = h;                      // mean over a single element
        float diff = h - mu;                 // == 0
        float var  = diff * diff;
        float y    = diff * rsqrtf(var + LN_EPS) * ln_gamma[0] + ln_beta[0];
        long  eo   = tbase + r;
        if (eo < nEdges) out[eo] = y;
    }
}

extern "C" void kernel_launch(void* const* d_in, const int* in_sizes, int n_in,
                              void* d_out, int out_size, void* d_ws, size_t ws_size,
                              hipStream_t stream) {
    const float*   x_node     = (const float*)d_in[0];
    const int64_t* edge_index = (const int64_t*)d_in[1];
    const float*   W1 = (const float*)d_in[2];
    const float*   b1 = (const float*)d_in[3];
    const float*   W2 = (const float*)d_in[4];
    const float*   b2 = (const float*)d_in[5];
    const float*   W3 = (const float*)d_in[6];
    const float*   b3 = (const float*)d_in[7];
    const float*   W4 = (const float*)d_in[8];
    const float*   b4 = (const float*)d_in[9];
    const float*   g  = (const float*)d_in[10];
    const float*   be = (const float*)d_in[11];
    float* out = (float*)d_out;

    const int nEdges = in_sizes[1] / 2;
    unsigned short* wt = (unsigned short*)d_ws;   // 8 KB of packed bf16 weights

    prep_weights<<<(8 * 512 + 255) / 256, 256, 0, stream>>>(W1, W2, W3, wt);

    const int tiles  = (nEdges + 15) / 16;
    const int blocks = (tiles + 7) / 8;           // 8 waves (tiles) per 256-thr block
    edge_mlp<<<blocks, 256, 0, stream>>>(x_node, edge_index, b1, b2, b3,
                                         W4, b4, g, be, wt, out, nEdges);
}
